// PoseHeadBlock_14843406975499
// MI455X (gfx1250) — matrix operand order repaired and checked by reading, not compiled, not run on hardware
//
#include <hip/hip_runtime.h>
#include <hip/hip_bf16.h>
#include <math.h>

// ---------------------------------------------------------------------------
// PoseHeadBlock on MI455X (gfx1250, wave32).
// Heavy GEMMs: v_wmma_f32_16x16x32_bf16, fp32 accumulate, LN fused on A-load,
// double-buffered LDS tiles (one barrier per K-step), fragment-major LDS
// layouts so every lane fragment is 2x ds_load_b128.
// ---------------------------------------------------------------------------

typedef __attribute__((ext_vector_type(16))) __bf16 v16bf;
typedef __attribute__((ext_vector_type(8)))  __bf16 v8bf;
typedef __attribute__((ext_vector_type(2)))  __bf16 v2bf;
typedef __attribute__((ext_vector_type(8)))  float  v8f;
typedef __attribute__((ext_vector_type(4)))  float  f4;

#define JQ 16      // query tokens
#define NH 16      // heads
#define HD 64      // head dim
#define DM 1024    // model dim
#define NSLOT 4
#define ATT_SCALE 0.125f   // 1/sqrt(64)

#define GF_GELU   1
#define GF_RESID  2
#define GF_LN     4
#define GF_MGUARD 8   // only for M not multiple of TM (FiLM GEMM)

#define TM 128
#define TN 128
#define TK 32

// ---------------------------------------------------------------------------
// Per-row mean / rstd for LayerNorm fusion.
// ---------------------------------------------------------------------------
__global__ __launch_bounds__(256) void k_rowstats(const float* __restrict__ X,
                                                  int ld, int D,
                                                  float* __restrict__ mean,
                                                  float* __restrict__ rstd) {
  __shared__ float s1[256];
  __shared__ float s2[256];
  const int row = blockIdx.x;
  const float* x = X + (size_t)row * ld;
  float a = 0.f, b = 0.f;
  for (int i = threadIdx.x * 4; i < D; i += 1024) {
    f4 v = *(const f4*)(x + i);
    a += v[0] + v[1] + v[2] + v[3];
    b += v[0] * v[0] + v[1] * v[1] + v[2] * v[2] + v[3] * v[3];
  }
  s1[threadIdx.x] = a; s2[threadIdx.x] = b;
  __syncthreads();
  for (int off = 128; off > 0; off >>= 1) {
    if ((int)threadIdx.x < off) {
      s1[threadIdx.x] += s1[threadIdx.x + off];
      s2[threadIdx.x] += s2[threadIdx.x + off];
    }
    __syncthreads();
  }
  if (threadIdx.x == 0) {
    float m = s1[0] / (float)D;
    float v = s2[0] / (float)D - m * m;
    mean[row] = m;
    rstd[row] = rsqrtf(v + 1e-5f);
  }
}

// ---------------------------------------------------------------------------
// FiLM apply: h' = h * (1 + s) + sh, s/sh from film[B, 2*DM].
// ---------------------------------------------------------------------------
__global__ __launch_bounds__(256) void k_film_apply(const float* __restrict__ h,
                                                    const float* __restrict__ film,
                                                    float* __restrict__ out) {
  const int idx = blockIdx.x * 256 + threadIdx.x;   // B*J*DM threads
  const int d = idx & (DM - 1);
  const int b = idx >> 14;                          // idx / (J*DM)
  const float s  = film[b * (2 * DM) + d];
  const float sh = film[b * (2 * DM) + DM + d];
  out[idx] = h[idx] * (1.f + s) + sh;
}

// ---------------------------------------------------------------------------
// Pack two [K,N] weight matrices side by side -> [K, 2N].
// ---------------------------------------------------------------------------
__global__ __launch_bounds__(256) void k_pack2(const float* __restrict__ Wa,
                                               const float* __restrict__ Wb,
                                               float* __restrict__ out,
                                               int N) {
  const int idx = blockIdx.x * 256 + threadIdx.x;   // K*2N threads
  const int row = idx / (2 * N);
  const int col = idx - row * 2 * N;
  out[idx] = (col < N) ? Wa[(size_t)row * N + col]
                       : Wb[(size_t)row * N + col - N];
}

// ---------------------------------------------------------------------------
// Tiled WMMA GEMM: C[M,N] = epilogue( LN?(A)[M,K] @ W[K,N] + bias )
//   Block tile 128x128, K-step 32, double-buffered LDS; 8 waves, each a
//   32x64 tile (2 A-frags x 4 B-frags = 8 WMMAs per K-step).
//   LDS layouts (fragment-major, each lane fragment = 32 contiguous bytes):
//     sA[row][k], natural k order (lane half h reads k 16h..16h+15)
//     sBt[col][k] (transposed)
//   Requires: N % 128 == 0, K % 32 == 0; M % 128 == 0 unless GF_MGUARD.
// ---------------------------------------------------------------------------
__global__ __launch_bounds__(256) void k_gemm(
    const float* __restrict__ A, int lda,
    const float* __restrict__ W, int ldw,
    const float* __restrict__ bias,
    const float* __restrict__ residual, int ldr,
    float* __restrict__ C, int ldc,
    int M, int N, int K,
    const float* __restrict__ ln_mean, const float* __restrict__ ln_rstd,
    const float* __restrict__ ln_g,    const float* __restrict__ ln_b,
    int flags) {
  __shared__ __bf16 sA[2][TM][TK];     // 2 x 8 KB
  __shared__ __bf16 sBt[2][TN][TK];    // 2 x 8 KB

  const int tid  = threadIdx.x;
  const int lane = tid & 31;
  const int wave = tid >> 5;           // 0..7
  const int wm   = wave >> 1;          // 0..3  (row group of 32)
  const int wn   = wave & 1;           // 0..1  (col group of 64)
  const int l16  = lane & 15;
  const int hi   = lane >> 4;          // 0 or 1

  const int blockM = blockIdx.y * TM;
  const int blockN = blockIdx.x * TN;

  // A loader: 2 threads per row, 16 consecutive k each
  const int ar   = tid >> 1;
  const int ac0  = (tid & 1) * 16;
  const int grow = blockM + ar;
  const int ap0  = (tid & 1) * 16;       // natural store pos of k-group 0
  const int ap1  = (tid & 1) * 16 + 8;   // natural store pos of k-group 1
  const bool useLN  = (flags & GF_LN) != 0;
  const bool mguard = (flags & GF_MGUARD) != 0;
  const bool arow_ok = !mguard || (grow < M);
  float lmean = 0.f, lrstd = 1.f;
  if (useLN && arow_ok) { lmean = ln_mean[grow]; lrstd = ln_rstd[grow]; }

  // B loader: thread covers 2 k-rows x 8 cols (coalesced global reads)
  const int bk  = (tid >> 4) * 2;        // 0,2,..,30
  const int bc0 = (tid & 15) * 8;        // 0..120

  v8f acc[2][4];
#pragma unroll
  for (int mi = 0; mi < 2; ++mi)
#pragma unroll
    for (int ni = 0; ni < 4; ++ni)
#pragma unroll
      for (int r = 0; r < 8; ++r) acc[mi][ni][r] = 0.f;

  f4 xa[4];            // staged A: 16 floats of one row
  f4 xb[4];            // staged B: 2 k-rows x 8 cols

  auto load_tile = [&](int k0) {
    if (arow_ok) {
      const f4* ap4 = (const f4*)(A + (size_t)grow * lda + k0 + ac0);
      __builtin_prefetch(A + (size_t)grow * lda + k0 + TK + ac0, 0, 1);
#pragma unroll
      for (int v = 0; v < 4; ++v) xa[v] = ap4[v];
      if (useLN) {
        const f4* g4 = (const f4*)(ln_g + k0 + ac0);
        const f4* b4 = (const f4*)(ln_b + k0 + ac0);
#pragma unroll
        for (int v = 0; v < 4; ++v)
          xa[v] = (xa[v] - lmean) * lrstd * g4[v] + b4[v];
      }
    } else {
#pragma unroll
      for (int v = 0; v < 4; ++v) xa[v] = (f4)0.f;
    }
    const float* wr = W + (size_t)(k0 + bk) * ldw + blockN + bc0;
    __builtin_prefetch(wr + (size_t)TK * ldw, 0, 1);
    xb[0] = *(const f4*)wr;
    xb[1] = *(const f4*)(wr + 4);
    xb[2] = *(const f4*)(wr + ldw);
    xb[3] = *(const f4*)(wr + ldw + 4);
  };

  auto store_tile = [&](int buf) {
    v8bf g0, g1;
#pragma unroll
    for (int e = 0; e < 8; ++e) {
      g0[e] = (__bf16)xa[e >> 2][e & 3];
      g1[e] = (__bf16)xa[2 + (e >> 2)][e & 3];
    }
    *(v8bf*)&sA[buf][ar][ap0] = g0;
    *(v8bf*)&sA[buf][ar][ap1] = g1;
#pragma unroll
    for (int i = 0; i < 8; ++i) {
      v2bf p;
      p[0] = (__bf16)((i < 4) ? xb[0][i] : xb[1][i - 4]);
      p[1] = (__bf16)((i < 4) ? xb[2][i] : xb[3][i - 4]);
      *(v2bf*)&sBt[buf][bc0 + i][bk] = p;
    }
  };

  // prologue: stage tile 0
  load_tile(0);
  store_tile(0);
  __syncthreads();

  int p = 0;
  for (int k0 = 0; k0 < K; k0 += TK) {
    const bool more = (k0 + TK) < K;
    if (more) load_tile(k0 + TK);   // global loads overlap this tile's math

    v16bf af[2], bfr[4];
#pragma unroll
    for (int mi = 0; mi < 2; ++mi)
      af[mi] = *(const v16bf*)&sA[p][wm * 32 + mi * 16 + l16][hi * 16];
#pragma unroll
    for (int ni = 0; ni < 4; ++ni)
      bfr[ni] = *(const v16bf*)&sBt[p][wn * 64 + ni * 16 + l16][hi * 16];

#pragma unroll
    for (int mi = 0; mi < 2; ++mi)
#pragma unroll
      for (int ni = 0; ni < 4; ++ni)
        acc[mi][ni] = __builtin_amdgcn_wmma_f32_16x16x32_bf16(
            false, af[mi], false, bfr[ni], (short)0, acc[mi][ni],
            false, false);

    if (more) store_tile(p ^ 1);
    __syncthreads();
    p ^= 1;
  }

  // ---- epilogue: lanes0-15 -> rows r, lanes16-31 -> rows 8+r; col = l16
#pragma unroll
  for (int mi = 0; mi < 2; ++mi) {
    const int rbase = blockM + wm * 32 + mi * 16 + hi * 8;
#pragma unroll
    for (int ni = 0; ni < 4; ++ni) {
      const int col = blockN + wn * 64 + ni * 16 + l16;
      const float bcol = bias[col];
#pragma unroll
      for (int r = 0; r < 8; ++r) {
        const int rr = rbase + r;
        if (!mguard || rr < M) {
          float v = acc[mi][ni][r] + bcol;
          if (flags & GF_RESID) v += residual[(size_t)rr * ldr + col];
          if (flags & GF_GELU)  v = 0.5f * v * (1.f + erff(v * 0.70710678118654752f));
          C[(size_t)rr * ldc + col] = v;
        }
      }
    }
  }
}

// ---------------------------------------------------------------------------
// Online-softmax attention: grid (NH, B); 16 queries, Lk keys, head dim 64.
// Kept fp32/VALU: ~1.5% of total FLOPs, preserves softmax precision.
// ---------------------------------------------------------------------------
__global__ __launch_bounds__(256) void k_attn(
    const float* __restrict__ Q, int ldq,
    const float* __restrict__ Kp, int ldk,
    const float* __restrict__ Vp, int ldv,
    float* __restrict__ O, int ldo,
    int Lk,
    const unsigned char* __restrict__ pa_mask,   // [B,J,Lk] or null
    const unsigned char* __restrict__ tok_mask,  // [B,Lk]   or null
    const int* __restrict__ slots,               // [B,Lk]
    const float* __restrict__ part_bias,         // [NH]
    const float* __restrict__ slot_bias) {       // [NH,NSLOT]
  __shared__ float sQ[JQ][HD];
  __shared__ float sK[16][HD];
  __shared__ float sV[16][HD];
  __shared__ float sP[JQ][16];
  __shared__ float sM[JQ];
  __shared__ float sS[JQ];
  __shared__ float sAl[JQ];

  const int b  = blockIdx.y;
  const int hh = blockIdx.x;
  const int t  = threadIdx.x;
  const int q  = t >> 4;         // 0..15
  const int li = t & 15;

  {
    const int d0 = li * 4;
    *(f4*)&sQ[q][d0] = *(const f4*)(Q + (size_t)(b * JQ + q) * ldq + hh * HD + d0);
  }
  if (t < JQ) { sM[t] = -__builtin_inff(); sS[t] = 0.f; }

  float o0 = 0.f, o1 = 0.f, o2 = 0.f, o3 = 0.f;

  for (int c0 = 0; c0 < Lk; c0 += 16) {
    __syncthreads();
    {
      const int d0 = li * 4;
      const size_t krow = (size_t)(b * Lk + c0 + q);
      *(f4*)&sK[q][d0] = *(const f4*)(Kp + krow * ldk + hh * HD + d0);
      *(f4*)&sV[q][d0] = *(const f4*)(Vp + krow * ldv + hh * HD + d0);
    }
    __syncthreads();

    float s = 0.f;
#pragma unroll
    for (int d = 0; d < HD; ++d) s += sQ[q][d] * sK[li][d];
    s *= ATT_SCALE;

    if (tok_mask) {
      const int l = c0 + li;
      if (pa_mask[((size_t)b * JQ + q) * Lk + l]) s += part_bias[hh];
      int sl = slots[(size_t)b * Lk + l];
      sl = sl < 0 ? 0 : (sl > NSLOT - 1 ? NSLOT - 1 : sl);
      s += slot_bias[hh * NSLOT + sl];
      if (!tok_mask[(size_t)b * Lk + l]) s = -3.4028234663852886e38f;
    }

    float m = s;
#pragma unroll
    for (int off = 1; off < 16; off <<= 1) m = fmaxf(m, __shfl_xor(m, off, 32));
    const float mOld = sM[q];
    const float mNew = fmaxf(mOld, m);
    const float pv = __expf(s - mNew);
    float psum = pv;
#pragma unroll
    for (int off = 1; off < 16; off <<= 1) psum += __shfl_xor(psum, off, 32);
    const float alpha = __expf(mOld - mNew);
    sP[q][li] = pv;
    if (li == 0) { sM[q] = mNew; sS[q] = sS[q] * alpha + psum; sAl[q] = alpha; }
    __syncthreads();

    const float al = sAl[q];
    const int dbase = li * 4;
    float a0 = 0.f, a1 = 0.f, a2 = 0.f, a3 = 0.f;
#pragma unroll
    for (int k = 0; k < 16; ++k) {
      const float pk = sP[q][k];
      a0 += pk * sV[k][dbase + 0];
      a1 += pk * sV[k][dbase + 1];
      a2 += pk * sV[k][dbase + 2];
      a3 += pk * sV[k][dbase + 3];
    }
    o0 = o0 * al + a0; o1 = o1 * al + a1; o2 = o2 * al + a2; o3 = o3 * al + a3;
  }

  __syncthreads();
  const float inv = 1.f / sS[q];
  float* op = O + (size_t)(b * JQ + q) * ldo + hh * HD + li * 4;
  op[0] = o0 * inv; op[1] = o1 * inv; op[2] = o2 * inv; op[3] = o3 * inv;
}

// ---------------------------------------------------------------------------
// Host launcher
// ---------------------------------------------------------------------------
static void launch_gemm(hipStream_t st,
                        const float* A, int lda, const float* W, int ldw,
                        const float* bias, const float* resid, int ldr,
                        float* C, int ldc, int M, int N, int K,
                        const float* lm, const float* lr,
                        const float* lg, const float* lb, int flags) {
  dim3 grid((N + TN - 1) / TN, (M + TM - 1) / TM);
  k_gemm<<<grid, 256, 0, st>>>(A, lda, W, ldw, bias, resid, ldr, C, ldc,
                               M, N, K, lm, lr, lg, lb, flags);
}

extern "C" void kernel_launch(void* const* d_in, const int* in_sizes, int n_in,
                              void* d_out, int out_size, void* d_ws, size_t ws_size,
                              hipStream_t stream) {
  const int B = 16, L = 2048, S = 256;
  const int MJ = B * JQ;          // 256
  const int ML = B * L;           // 32768
  const int MS = B * S;           // 4096

  const float* h_in      = (const float*)d_in[0];
  const float* cond      = (const float*)d_in[1];
  const float* token_kv  = (const float*)d_in[2];
  const unsigned char* tok_mask = (const unsigned char*)d_in[3];
  const unsigned char* pa_mask  = (const unsigned char*)d_in[4];
  const int*   slots     = (const int*)d_in[5];
  const float* spatial   = (const float*)d_in[6];
  const float* film_w    = (const float*)d_in[7];
  const float* film_b    = (const float*)d_in[8];
  const float* ln_g      = (const float*)d_in[9];
  const float* ln_b      = (const float*)d_in[10];
  const float* sa_in_w   = (const float*)d_in[11];
  const float* sa_in_b   = (const float*)d_in[12];
  const float* sa_out_w  = (const float*)d_in[13];
  const float* sa_out_b  = (const float*)d_in[14];
  const float* cq_w = (const float*)d_in[15]; const float* cq_b = (const float*)d_in[16];
  const float* ck_w = (const float*)d_in[17]; const float* ck_b = (const float*)d_in[18];
  const float* cv_w = (const float*)d_in[19]; const float* cv_b = (const float*)d_in[20];
  const float* co_w = (const float*)d_in[21]; const float* co_b = (const float*)d_in[22];
  const float* sp_in_w   = (const float*)d_in[23];
  const float* sp_in_b   = (const float*)d_in[24];
  const float* sp_out_w  = (const float*)d_in[25];
  const float* sp_out_b  = (const float*)d_in[26];
  const float* part_bias = (const float*)d_in[27];
  const float* slot_bias = (const float*)d_in[28];
  const float* ffn_w1 = (const float*)d_in[29]; const float* ffn_b1 = (const float*)d_in[30];
  const float* ffn_w2 = (const float*)d_in[31]; const float* ffn_b2 = (const float*)d_in[32];

  // ---- workspace bump allocation (floats) ----
  float* ws = (float*)d_ws;
  size_t off = 0;
  auto alloc = [&](size_t n) { float* p = ws + off; off += n; return p; };
  float* film    = alloc((size_t)B * 2 * DM);
  float* hbuf    = alloc((size_t)MJ * DM);
  float* qkv     = alloc((size_t)MJ * 3 * DM);
  float* attn_o  = alloc((size_t)MJ * DM);
  float* qbuf    = alloc((size_t)MJ * DM);
  float* wpack   = alloc((size_t)DM * 2 * DM);      // [ck_w | cv_w] and bias pair
  float* bpack   = alloc((size_t)2 * DM);
  float* statm_h = alloc(MJ); float* stats_h = alloc(MJ);
  float* statm_k = alloc(ML); float* stats_k = alloc(ML);
  float* statm_s = alloc(MS); float* stats_s = alloc(MS);
  float* kvbuf   = alloc((size_t)ML * 2 * DM);      // packed K|V projections
  float* kvsp    = kvbuf;                           // reused after token CA
  float* midbuf  = kvbuf + (size_t)MS * 2 * DM;     // reused after token CA
  float* outp    = (float*)d_out;

  // 1) FiLM: film = cond @ film_w + film_b   (M=16, N=2048, K=512)
  launch_gemm(stream, cond, 512, film_w, 2 * DM, film_b, nullptr, 0,
              film, 2 * DM, B, 2 * DM, 512, nullptr, nullptr, nullptr, nullptr,
              GF_MGUARD);
  k_film_apply<<<(MJ * DM) / 256, 256, 0, stream>>>(h_in, film, hbuf);

  // 2) Self-attention
  k_rowstats<<<MJ, 256, 0, stream>>>(hbuf, DM, DM, statm_h, stats_h);
  launch_gemm(stream, hbuf, DM, sa_in_w, 3 * DM, sa_in_b, nullptr, 0,
              qkv, 3 * DM, MJ, 3 * DM, DM,
              statm_h, stats_h, ln_g + 0 * DM, ln_b + 0 * DM, GF_LN);
  k_attn<<<dim3(NH, B), 256, 0, stream>>>(qkv, 3 * DM, qkv + DM, 3 * DM,
                                          qkv + 2 * DM, 3 * DM, attn_o, DM, JQ,
                                          nullptr, nullptr, nullptr, nullptr, nullptr);
  launch_gemm(stream, attn_o, DM, sa_out_w, DM, sa_out_b, hbuf, DM,
              hbuf, DM, MJ, DM, DM, nullptr, nullptr, nullptr, nullptr, GF_RESID);

  // 3) Masked token cross-attention; K|V fused into one 32768x2048x1024 GEMM
  k_pack2<<<(DM * 2 * DM) / 256, 256, 0, stream>>>(ck_w, cv_w, wpack, DM);
  k_pack2<<<(2 * DM + 255) / 256, 256, 0, stream>>>(ck_b, cv_b, bpack, DM);
  k_rowstats<<<MJ, 256, 0, stream>>>(hbuf, DM, DM, statm_h, stats_h);
  k_rowstats<<<ML, 256, 0, stream>>>(token_kv, DM, DM, statm_k, stats_k);
  launch_gemm(stream, hbuf, DM, cq_w, DM, cq_b, nullptr, 0, qbuf, DM,
              MJ, DM, DM, statm_h, stats_h, ln_g + 1 * DM, ln_b + 1 * DM, GF_LN);
  launch_gemm(stream, token_kv, DM, wpack, 2 * DM, bpack, nullptr, 0, kvbuf, 2 * DM,
              ML, 2 * DM, DM, statm_k, stats_k, ln_g + 2 * DM, ln_b + 2 * DM, GF_LN);
  k_attn<<<dim3(NH, B), 256, 0, stream>>>(qbuf, DM, kvbuf, 2 * DM, kvbuf + DM, 2 * DM,
                                          attn_o, DM, L,
                                          pa_mask, tok_mask, slots, part_bias, slot_bias);
  launch_gemm(stream, attn_o, DM, co_w, DM, co_b, hbuf, DM,
              hbuf, DM, MJ, DM, DM, nullptr, nullptr, nullptr, nullptr, GF_RESID);

  // 4) Spatial cross-attention (sp_in_w already has K|V packed at cols DM..3DM)
  k_rowstats<<<MJ, 256, 0, stream>>>(hbuf, DM, DM, statm_h, stats_h);
  k_rowstats<<<MS, 256, 0, stream>>>(spatial, DM, DM, statm_s, stats_s);
  launch_gemm(stream, hbuf, DM, sp_in_w, 3 * DM, sp_in_b, nullptr, 0, qbuf, DM,
              MJ, DM, DM, statm_h, stats_h, ln_g + 3 * DM, ln_b + 3 * DM, GF_LN);
  launch_gemm(stream, spatial, DM, sp_in_w + DM, 3 * DM, sp_in_b + DM, nullptr, 0,
              kvsp, 2 * DM, MS, 2 * DM, DM,
              statm_s, stats_s, ln_g + 4 * DM, ln_b + 4 * DM, GF_LN);
  k_attn<<<dim3(NH, B), 256, 0, stream>>>(qbuf, DM, kvsp, 2 * DM, kvsp + DM, 2 * DM,
                                          attn_o, DM, S,
                                          nullptr, nullptr, nullptr, nullptr, nullptr);
  launch_gemm(stream, attn_o, DM, sp_out_w, DM, sp_out_b, hbuf, DM,
              hbuf, DM, MJ, DM, DM, nullptr, nullptr, nullptr, nullptr, GF_RESID);

  // 5) FFN with exact-erf GELU
  k_rowstats<<<MJ, 256, 0, stream>>>(hbuf, DM, DM, statm_h, stats_h);
  launch_gemm(stream, hbuf, DM, ffn_w1, 4 * DM, ffn_b1, nullptr, 0, midbuf, 4 * DM,
              MJ, 4 * DM, DM, statm_h, stats_h, ln_g + 5 * DM, ln_b + 5 * DM,
              GF_LN | GF_GELU);
  launch_gemm(stream, midbuf, 4 * DM, ffn_w2, DM, ffn_b2, hbuf, DM,
              outp, DM, MJ, DM, 4 * DM, nullptr, nullptr, nullptr, nullptr, GF_RESID);

  (void)in_sizes; (void)n_in; (void)out_size; (void)ws_size;
}
